// LSTM_23862838296993
// MI455X (gfx1250) — matrix operand order repaired and checked by reading, not compiled, not run on hardware
//
#include <hip/hip_runtime.h>
#include <hip/hip_bf16.h>

typedef __attribute__((ext_vector_type(16))) __bf16 v16bf;
typedef __attribute__((ext_vector_type(8)))  __bf16 bf16x8;
typedef __attribute__((ext_vector_type(8)))  float  v8f;

#define BDIM 4096
#define DDIM 1024     // == HDIM, x-phase and h-phase have identical shapes
#define HDIM 1024
#define MT 128        // batch-tile
#define NT 64         // hidden-tile
#define KT 32         // K-step (matches 16x16x32 bf16 WMMA)

// pack two fp32 -> two bf16 (round-to-nearest) in ONE v_perm_b32:
// round via +0x8000 carry into the high half, then byte-select the two
// high halves: out[15:0]=bf16(a), out[31:16]=bf16(b).
__device__ __forceinline__ unsigned pk2(float a, float b) {
    unsigned ua = __builtin_bit_cast(unsigned, a) + 0x8000u;
    unsigned ub = __builtin_bit_cast(unsigned, b) + 0x8000u;
    return __builtin_amdgcn_perm(ub, ua, 0x07060302u);
}

__device__ __forceinline__ uint4 cvt8(float4 a, float4 b) {
    uint4 r;
    r.x = pk2(a.x, a.y); r.y = pk2(a.z, a.w);
    r.z = pk2(b.x, b.y); r.w = pk2(b.z, b.w);
    return r;
}

__device__ __forceinline__ uint2 cvt4(float4 a) {
    uint2 r;
    r.x = pk2(a.x, a.y); r.y = pk2(a.z, a.w);
    return r;
}

__global__ __launch_bounds__(512)
void LSTM_wmma_kernel(const float* __restrict__ hh,   // hidden_h [B,H]
                      const float* __restrict__ hc,   // hidden_c [B,H]
                      const float* __restrict__ x,    // x  [B,D]
                      const float* __restrict__ Wx,   // Wx [G,H,D]
                      const float* __restrict__ Wh,   // Wh [G,H,H]
                      const float* __restrict__ bias, // bias [G,H]
                      float* __restrict__ out)        // [out_h | out_c], each [B,H]
{
    __shared__ __align__(16) __bf16 At[MT * KT];      // 128 x 32 bf16 (A tile)
    __shared__ __align__(16) __bf16 Bt[4][NT * KT];   // 4 gates x 64 x 32 bf16

    const int tid  = threadIdx.x;
    const int lane = tid & 31;
    const int wave = tid >> 5;          // 0..15
    const int wm   = wave >> 2;         // 0..3 : M-group of this wave
    const int wn   = wave & 3;          // 0..3 : N-subtile of this wave
    const int b0   = blockIdx.y * MT;
    const int n0   = blockIdx.x * NT;

    v8f acc[2][4];
#pragma unroll
    for (int s = 0; s < 2; ++s)
#pragma unroll
        for (int g = 0; g < 4; ++g) acc[s][g] = (v8f)(0.0f);

    // staging decomposition (512 threads):
    //   A: 128 rows x 32 cols fp32, 8 floats/thread
    //   B: per gate 64 rows x 32 cols fp32, 4 floats/thread
    const int arow = tid >> 2;          // 0..127
    const int acol = (tid & 3) * 8;     // 0,8,16,24
    const int brow = tid >> 3;          // 0..63
    const int bcol = (tid & 7) * 4;     // 0,4,...,28

    // fragment addressing (wave32 CDNA5 WMMA layouts)
    const int mrow  = lane & 15;
    const int khalf = lane >> 4;        // K half-select (and C-layout M offset)
    const int ncol  = lane & 15;

    // two branch-free phases: (x, Wx) then (hidden_h, Wh); same [.,1024] shapes
#pragma unroll 1
    for (int phase = 0; phase < 2; ++phase) {
        const float* Ab = phase ? hh : x;      // [B, 1024]
        const float* Wb = phase ? Wh : Wx;     // [G*H, 1024]

#pragma unroll 1
        for (int k0 = 0; k0 < DDIM; k0 += KT) {
            // ---- global -> registers (issued before barrier; overlaps MMAs)
            float4 av0, av1, bv[4];
            {
                const float* asrc = Ab + (size_t)(b0 + arow) * DDIM + k0 + acol;
                av0 = *(const float4*)(asrc + 0);
                av1 = *(const float4*)(asrc + 4);
#pragma unroll
                for (int g = 0; g < 4; ++g) {
                    const float* bsrc =
                        Wb + (size_t)(g * HDIM + n0 + brow) * DDIM + k0 + bcol;
                    bv[g] = *(const float4*)bsrc;
                }
            }
            __syncthreads();    // previous iteration finished reading LDS
            // ---- registers -> LDS as bf16 (v_perm packing, 16B/8B ds stores)
            *(uint4*)&At[arow * KT + acol] = cvt8(av0, av1);
#pragma unroll
            for (int g = 0; g < 4; ++g)
                *(uint2*)&Bt[g][brow * KT + bcol] = cvt4(bv[g]);
            __syncthreads();

            // ---- load fragments and issue WMMAs
            v16bf afrag[2];
#pragma unroll
            for (int s = 0; s < 2; ++s) {
                // A 16x32 bf16 layout: lane<16 holds M=lane, K in {0..7}u{16..23};
                // lane>=16 holds M=lane-16, K in {8..15}u{24..31}.
                const __bf16* ap =
                    &At[(wm * 16 + s * 64 + mrow) * KT + khalf * 8];
                bf16x8 lo = *(const bf16x8*)(ap);        // K = khalf*8 ..
                bf16x8 hi = *(const bf16x8*)(ap + 16);   // K = 16+khalf*8 ..
                afrag[s] = __builtin_shufflevector(lo, hi,
                            0,1,2,3,4,5,6,7,8,9,10,11,12,13,14,15);
            }
#pragma unroll
            for (int g = 0; g < 4; ++g) {
                // B 32x16 bf16: lanes 0-15 hold N=lane, K in {0..7}u{16..23};
                // lanes 16-31 hold N=lane-16, K in {8..15}u{24..31}.
                const __bf16* bp =
                    &Bt[g][(wn * 16 + ncol) * KT + khalf * 8];
                bf16x8 blo = *(const bf16x8*)(bp);
                bf16x8 bhi = *(const bf16x8*)(bp + 16);
                v16bf bfrag = __builtin_shufflevector(blo, bhi,
                            0,1,2,3,4,5,6,7,8,9,10,11,12,13,14,15);
                acc[0][g] = __builtin_amdgcn_wmma_f32_16x16x32_bf16(
                    false, afrag[0], false, bfrag, (short)0, acc[0][g], false, false);
                acc[1][g] = __builtin_amdgcn_wmma_f32_16x16x32_bf16(
                    false, afrag[1], false, bfrag, (short)0, acc[1][g], false, false);
            }
        }
    }

    // ---- epilogue: bias + LSTM gating, straight from C-fragment layout
    const int nglob = n0 + wn * 16 + ncol;
    float bv4[4];
#pragma unroll
    for (int g = 0; g < 4; ++g) bv4[g] = bias[g * HDIM + nglob];

    float* outH = out;
    float* outC = out + (size_t)BDIM * HDIM;

#pragma unroll
    for (int s = 0; s < 2; ++s) {
#pragma unroll
        for (int r = 0; r < 8; ++r) {
            // C/D layout: VGPR r -> M = r (lanes 0-15), r+8 (lanes 16-31)
            const int bglob  = b0 + wm * 16 + s * 64 + khalf * 8 + r;
            const size_t idx = (size_t)bglob * HDIM + nglob;
            const float pf = acc[s][0][r] + bv4[0];
            const float pi = acc[s][1][r] + bv4[1];
            const float po = acc[s][2][r] + bv4[2];
            const float pc = acc[s][3][r] + bv4[3];
            const float fg = 1.0f / (1.0f + __expf(-pf));
            const float ig = 1.0f / (1.0f + __expf(-pi));
            const float og = 1.0f / (1.0f + __expf(-po));
            const float e2  = __expf(2.0f * pc);
            const float cin = (e2 - 1.0f) / (e2 + 1.0f);     // tanh(pre_c)
            const float cprev = hc[idx];
            const float cnew  = fg * cprev + ig * cin;
            const float e2c = __expf(2.0f * cnew);
            const float tc  = (e2c - 1.0f) / (e2c + 1.0f);   // tanh(c_new)
            outH[idx] = og * tc;
            outC[idx] = cnew;
        }
    }
}

extern "C" void kernel_launch(void* const* d_in, const int* in_sizes, int n_in,
                              void* d_out, int out_size, void* d_ws, size_t ws_size,
                              hipStream_t stream) {
    const float* hh   = (const float*)d_in[0];  // hidden_h
    const float* hc   = (const float*)d_in[1];  // hidden_c
    const float* x    = (const float*)d_in[2];  // x
    const float* Wx   = (const float*)d_in[3];  // Wx
    const float* Wh   = (const float*)d_in[4];  // Wh
    const float* bias = (const float*)d_in[5];  // bias
    float* out = (float*)d_out;                 // [out_h | out_c]

    dim3 grid(HDIM / NT, BDIM / MT);            // (16, 32)
    dim3 block(512);
    hipLaunchKernelGGL(LSTM_wmma_kernel, grid, block, 0, stream,
                       hh, hc, x, Wx, Wh, bias, out);
}
